// TransformerLayer_52158082842772
// MI455X (gfx1250) — matrix operand order repaired and checked by reading, not compiled, not run on hardware
//
#include <hip/hip_runtime.h>
#include <hip/hip_bf16.h>

#define LEN_PFX 513
#define S_LEN   2048
#define D_DIM   256
#define H_CNT   8
#define HD_DIM  32
#define FF_DIM  1024
#define B_CNT   4
#define NTOK    (B_CNT * S_LEN)   // 8192 tokens

typedef __attribute__((ext_vector_type(16))) _Float16 v16h;
typedef __attribute__((ext_vector_type(8)))  _Float16 v8h;
typedef __attribute__((ext_vector_type(8)))  float    v8f;

union H16 { v16h v; v8h h[2]; };

__device__ __forceinline__ v8f wmma_f16(const H16& a, const H16& b, v8f c) {
  // D = A(16x32 f16) * B(32x16 f16) + C(16x16 f32)
  return __builtin_amdgcn_wmma_f32_16x16x32_f16(false, a.v, false, b.v,
                                                (short)0, c, false, false);
}

// ---------------------------------------------------------------------------
// fp32 -> fp16 conversion
// ---------------------------------------------------------------------------
__global__ void cvt_f32_f16(const float* __restrict__ in, _Float16* __restrict__ out, int n) {
  int i = blockIdx.x * blockDim.x + threadIdx.x;
  if (i < n) out[i] = (_Float16)in[i];
}

// ---------------------------------------------------------------------------
// Generic GEMM: out = epilogue( A[M,K](f16) @ W[N,K]^T(f16) + bias[N] )
// EP 0: fp32 store, row-major [M,N]
// EP 1: exact GELU, f16 store, row-major [M,N]
// EP 2: f16 store, head-split [B,H,S,HD], value scaled by `scale`
// EP 3: f16 store, head-split transposed [B,H,HD,S] (packed b128 stores)
// ---------------------------------------------------------------------------
template <int EP>
__global__ __launch_bounds__(256)
void gemm_xwt(const _Float16* __restrict__ A, const _Float16* __restrict__ W,
              const float* __restrict__ bias, void* __restrict__ outp,
              int M, int N, int K, float scale) {
  __shared__ alignas(16) _Float16 As[128][40];  // BM x BK, +8 pad (16B)
  __shared__ alignas(16) _Float16 Bs[64][40];   // BN x BK, +8 pad

  const int t     = threadIdx.x;
  const int lane  = t & 31;
  const int wave  = t >> 5;            // 8 waves
  const int nl    = lane & 15;
  const int khalf = lane >> 4;
  const int waveM = wave >> 1;         // 0..3 -> 32-row strip
  const int waveN = wave & 1;          // 0..1 -> 32-col strip
  const int rowBlk = blockIdx.y * 128;
  const int colBlk = blockIdx.x * 64;

  // cooperative tile loads: A 128x32 (2 thr/row, 16 halfs each), B 64x32
  const int ar = t >> 1, ac = (t & 1) * 16;
  const int br = t >> 2, bc = (t & 3) * 8;

  v8f acc[2][2] = {};

  for (int k0 = 0; k0 < K; k0 += 32) {
    const v8h* ag = (const v8h*)(A + (size_t)(rowBlk + ar) * K + k0 + ac);
    v8h a0 = ag[0];
    v8h a1 = ag[1];
    v8h b0 = *(const v8h*)(W + (size_t)(colBlk + br) * K + k0 + bc);
    if (k0 + 32 < K)
      __builtin_prefetch(A + (size_t)(rowBlk + ar) * K + k0 + 32 + ac, 0, 3);

    __syncthreads();
    *(v8h*)&As[ar][ac]     = a0;
    *(v8h*)&As[ar][ac + 8] = a1;
    *(v8h*)&Bs[br][bc]     = b0;
    __syncthreads();

    H16 af[2], bf[2];
#pragma unroll
    for (int i = 0; i < 2; ++i) {  // A frag: lanes 0-15 K{0-7,16-23}, lanes 16-31 K{8-15,24-31}
      const _Float16* ap = &As[waveM * 32 + i * 16 + nl][0];
      af[i].h[0] = *(const v8h*)(ap + khalf * 8);
      af[i].h[1] = *(const v8h*)(ap + khalf * 8 + 16);
    }
#pragma unroll
    for (int j = 0; j < 2; ++j) {  // B frag: lanes 0-15 K{0-7,16-23}, lanes 16-31 K{8-15,24-31}
      const _Float16* bp = &Bs[waveN * 32 + j * 16 + nl][khalf * 8];
      bf[j].h[0] = *(const v8h*)bp;
      bf[j].h[1] = *(const v8h*)(bp + 16);
    }
#pragma unroll
    for (int i = 0; i < 2; ++i)
#pragma unroll
      for (int j = 0; j < 2; ++j)
        acc[i][j] = wmma_f16(af[i], bf[j], acc[i][j]);
  }

  // epilogue: C frag element (v,lane) -> row = v + 8*(lane>>4), col = lane&15
#pragma unroll
  for (int i = 0; i < 2; ++i) {
#pragma unroll
    for (int j = 0; j < 2; ++j) {
      const int col  = colBlk + waveN * 32 + j * 16 + nl;
      const float bv = bias[col];
      if (EP == 3) {
        // rows for v=0..7 are consecutive s -> pack one b128 store
        const int rbase = rowBlk + waveM * 32 + i * 16 + 8 * khalf;
        const int b = rbase >> 11, s = rbase & 2047;
        const int h = col >> 5, hd = col & 31;
        v8h pk;
#pragma unroll
        for (int v = 0; v < 8; ++v) pk[v] = (_Float16)(acc[i][j][v] + bv);
        *(v8h*)&((_Float16*)outp)[(((size_t)(b * H_CNT + h)) * HD_DIM + hd) * S_LEN + s] = pk;
      } else {
#pragma unroll
        for (int v = 0; v < 8; ++v) {
          const int row = rowBlk + waveM * 32 + i * 16 + v + 8 * khalf;
          float val = acc[i][j][v] + bv;
          if (EP == 0) {
            ((float*)outp)[(size_t)row * N + col] = val;
          } else if (EP == 1) {
            float g = 0.5f * val * (1.0f + erff(val * 0.70710678118654752f));
            ((_Float16*)outp)[(size_t)row * N + col] = (_Float16)g;
          } else {  // EP == 2
            val *= scale;
            const int b = row >> 11, s = row & 2047;
            const int h = col >> 5,  hd = col & 31;
            ((_Float16*)outp)[(((size_t)(b * H_CNT + h)) * S_LEN + s) * HD_DIM + hd] = (_Float16)val;
          }
        }
      }
    }
  }
}

// ---------------------------------------------------------------------------
// Flash attention: per wave, 16 query rows; online softmax; WMMA QK^T, P@V,
// and WMMA-based row-sum (P @ ones broadcasts each row sum to all lanes).
// Q pre-scaled by 1/sqrt(HD). K: [B,H,S,HD] f16, V: [B,H,HD,S] f16 (transposed).
// Mask: j > i  ||  (i < LEN_PFX && j >= LEN_PFX)
// ---------------------------------------------------------------------------
__global__ __launch_bounds__(256)
void attn_kernel(const _Float16* __restrict__ Q, const _Float16* __restrict__ K,
                 const _Float16* __restrict__ Vt, _Float16* __restrict__ AO) {
  __shared__ alignas(16) _Float16 Plds[8][16][40];  // per-wave P staging

  const int t     = threadIdx.x;
  const int lane  = t & 31;
  const int wave  = t >> 5;
  const int nl    = lane & 15;
  const int khalf = lane >> 4;
  const int bh    = blockIdx.y;                 // 0..31
  const int q0    = blockIdx.x * 128 + wave * 16;

  const _Float16* Qp = Q  + (size_t)bh * S_LEN * HD_DIM;
  const _Float16* Kp = K  + (size_t)bh * S_LEN * HD_DIM;
  const _Float16* Vp = Vt + (size_t)bh * HD_DIM * S_LEN;

  H16 qf;  // A frag: 16 rows x HD(=32) dims, loaded once
  {
    const _Float16* qr = Qp + (size_t)(q0 + nl) * HD_DIM;
    qf.h[0] = *(const v8h*)(qr + khalf * 8);
    qf.h[1] = *(const v8h*)(qr + khalf * 8 + 16);
  }

  H16 onesf;  // B frag of all-ones for WMMA row-sum
#pragma unroll
  for (int i = 0; i < 16; ++i) onesf.v[i] = (_Float16)1.0f;

  v8f o0 = {}, o1 = {};          // O accumulator 16x32
  float mr[8], ls[8], al[8];
#pragma unroll
  for (int v = 0; v < 8; ++v) { mr[v] = -1e30f; ls[v] = 0.0f; }

  const int jend = (q0 + 15) / 32 + 1;  // causal bound over 32-key tiles
  for (int j = 0; j < jend; ++j) {
    const int kbase = j * 32;

    H16 kf0, kf1;  // B frags for K^T: table order per key row
    {
      const _Float16* kr = Kp + (size_t)(kbase + nl) * HD_DIM + khalf * 8;
      kf0.h[0] = *(const v8h*)kr;
      kf0.h[1] = *(const v8h*)(kr + 16);
      const _Float16* kr2 = Kp + (size_t)(kbase + 16 + nl) * HD_DIM + khalf * 8;
      kf1.h[0] = *(const v8h*)kr2;
      kf1.h[1] = *(const v8h*)(kr2 + 16);
    }
    v8f zc = {};
    v8f s0 = wmma_f16(qf, kf0, zc);
    v8f s1 = wmma_f16(qf, kf1, zc);

#pragma unroll
    for (int v = 0; v < 8; ++v) {
      const int qg  = q0 + v + 8 * khalf;
      const int kg0 = kbase + nl;
      const int kg1 = kbase + 16 + nl;
      if (kg0 > qg || (qg < LEN_PFX && kg0 >= LEN_PFX)) s0[v] = -1e30f;
      if (kg1 > qg || (qg < LEN_PFX && kg1 >= LEN_PFX)) s1[v] = -1e30f;

      float mv = fmaxf(s0[v], s1[v]);
#pragma unroll
      for (int off = 8; off >= 1; off >>= 1) mv = fmaxf(mv, __shfl_xor(mv, off, 32));
      const float nm = fmaxf(mr[v], mv);
      al[v] = __expf(mr[v] - nm);
      mr[v] = nm;
      s0[v] = __expf(s0[v] - nm);
      s1[v] = __expf(s1[v] - nm);
      o0[v] *= al[v];
      o1[v] *= al[v];
    }

    // C-layout -> A-layout for P via per-wave LDS round trip
#pragma unroll
    for (int v = 0; v < 8; ++v) {
      const int m = v + 8 * khalf;
      Plds[wave][m][nl]      = (_Float16)s0[v];
      Plds[wave][m][nl + 16] = (_Float16)s1[v];
    }
    H16 pf;
    {
      const _Float16* pr = &Plds[wave][nl][0];
      pf.h[0] = *(const v8h*)(pr + khalf * 8);
      pf.h[1] = *(const v8h*)(pr + khalf * 8 + 16);
    }

    // WMMA row-sum: rs[v] = sum_k P[row][k], broadcast to all 16 lanes
    v8f rs = wmma_f16(pf, onesf, zc);
#pragma unroll
    for (int v = 0; v < 8; ++v) ls[v] = ls[v] * al[v] + rs[v];

    H16 vf0, vf1;  // B frags for V: table order, V is stored [HD,S]
    {
      const _Float16* vr = Vp + (size_t)nl * S_LEN + kbase + khalf * 8;
      vf0.h[0] = *(const v8h*)vr;
      vf0.h[1] = *(const v8h*)(vr + 16);
      const _Float16* vr2 = Vp + (size_t)(16 + nl) * S_LEN + kbase + khalf * 8;
      vf1.h[0] = *(const v8h*)vr2;
      vf1.h[1] = *(const v8h*)(vr2 + 16);
    }
    o0 = wmma_f16(pf, vf0, o0);
    o1 = wmma_f16(pf, vf1, o1);
  }

#pragma unroll
  for (int v = 0; v < 8; ++v) {
    const float inv = 1.0f / ls[v];
    const int qg = q0 + v + 8 * khalf;
    const int b  = bh >> 3, h = bh & 7;
    const size_t tok = (size_t)b * S_LEN + qg;
    AO[tok * D_DIM + h * HD_DIM + nl]      = (_Float16)(o0[v] * inv);
    AO[tok * D_DIM + h * HD_DIM + 16 + nl] = (_Float16)(o1[v] * inv);
  }
}

// ---------------------------------------------------------------------------
// LayerNorm(a + b) * g + be. One wave per 256-wide row; population variance.
// ---------------------------------------------------------------------------
__global__ __launch_bounds__(256)
void ln_residual(const float* __restrict__ a, const float* __restrict__ bvec,
                 const float* __restrict__ g, const float* __restrict__ be,
                 float* __restrict__ out32, _Float16* __restrict__ out16) {
  const int lane = threadIdx.x & 31;
  const int wave = threadIdx.x >> 5;
  const size_t row  = (size_t)blockIdx.x * 8 + wave;
  const size_t base = row * D_DIM + lane * 8;

  float x[8];
#pragma unroll
  for (int i = 0; i < 8; ++i) x[i] = a[base + i] + bvec[base + i];
  float s = 0.f, ss = 0.f;
#pragma unroll
  for (int i = 0; i < 8; ++i) { s += x[i]; ss += x[i] * x[i]; }
#pragma unroll
  for (int off = 16; off >= 1; off >>= 1) {
    s  += __shfl_xor(s, off, 32);
    ss += __shfl_xor(ss, off, 32);
  }
  const float m    = s * (1.0f / 256.0f);
  const float var  = ss * (1.0f / 256.0f) - m * m;
  const float rstd = rsqrtf(var + 1e-5f);
#pragma unroll
  for (int i = 0; i < 8; ++i) {
    const int c   = lane * 8 + i;
    const float y = (x[i] - m) * rstd * g[c] + be[c];
    out32[row * D_DIM + c] = y;
    if (out16) out16[row * D_DIM + c] = (_Float16)y;
  }
}

// ---------------------------------------------------------------------------
extern "C" void kernel_launch(void* const* d_in, const int* in_sizes, int n_in,
                              void* d_out, int out_size, void* d_ws, size_t ws_size,
                              hipStream_t stream) {
  (void)in_sizes; (void)n_in; (void)out_size; (void)ws_size;

  const float* src = (const float*)d_in[0];
  const float* Wq  = (const float*)d_in[1];
  const float* bq  = (const float*)d_in[2];
  const float* Wk  = (const float*)d_in[3];
  const float* bk  = (const float*)d_in[4];
  const float* Wv  = (const float*)d_in[5];
  const float* bv  = (const float*)d_in[6];
  const float* Wo  = (const float*)d_in[7];
  const float* bo  = (const float*)d_in[8];
  const float* W1  = (const float*)d_in[9];
  const float* b1  = (const float*)d_in[10];
  const float* W2  = (const float*)d_in[11];
  const float* b2  = (const float*)d_in[12];
  const float* g1  = (const float*)d_in[13];
  const float* be1 = (const float*)d_in[14];
  const float* g2  = (const float*)d_in[15];
  const float* be2 = (const float*)d_in[16];
  float* out = (float*)d_out;

  char* ws = (char*)d_ws;
  size_t off = 0;
  auto take = [&](size_t bytes) -> void* {
    void* p = ws + off;
    off = (off + bytes + 255) & ~(size_t)255;
    return p;
  };

  _Float16* xh    = (_Float16*)take((size_t)NTOK * D_DIM * 2);
  _Float16* Wqh   = (_Float16*)take((size_t)D_DIM * D_DIM * 2);
  _Float16* Wkh   = (_Float16*)take((size_t)D_DIM * D_DIM * 2);
  _Float16* Wvh   = (_Float16*)take((size_t)D_DIM * D_DIM * 2);
  _Float16* Woh   = (_Float16*)take((size_t)D_DIM * D_DIM * 2);
  _Float16* W1h   = (_Float16*)take((size_t)FF_DIM * D_DIM * 2);
  _Float16* W2h   = (_Float16*)take((size_t)D_DIM * FF_DIM * 2);
  _Float16* Qh    = (_Float16*)take((size_t)NTOK * D_DIM * 2);   // [B,H,S,HD]
  _Float16* Kh    = (_Float16*)take((size_t)NTOK * D_DIM * 2);   // [B,H,S,HD]
  _Float16* Vth   = (_Float16*)take((size_t)NTOK * D_DIM * 2);   // [B,H,HD,S]
  _Float16* AOh   = (_Float16*)take((size_t)NTOK * D_DIM * 2);   // [tok, D]
  float*    Y     = (float*)   take((size_t)NTOK * D_DIM * 4);   // GEMM fp32 out (reused)
  float*    src2  = (float*)   take((size_t)NTOK * D_DIM * 4);
  _Float16* src2h = (_Float16*)take((size_t)NTOK * D_DIM * 2);
  _Float16* Hh    = (_Float16*)take((size_t)NTOK * FF_DIM * 2);

  auto cdiv = [](int a, int b) { return (a + b - 1) / b; };

  // --- fp32 -> fp16 staging ---
  cvt_f32_f16<<<cdiv(NTOK * D_DIM, 256), 256, 0, stream>>>(src, xh, NTOK * D_DIM);
  cvt_f32_f16<<<cdiv(D_DIM * D_DIM, 256), 256, 0, stream>>>(Wq, Wqh, D_DIM * D_DIM);
  cvt_f32_f16<<<cdiv(D_DIM * D_DIM, 256), 256, 0, stream>>>(Wk, Wkh, D_DIM * D_DIM);
  cvt_f32_f16<<<cdiv(D_DIM * D_DIM, 256), 256, 0, stream>>>(Wv, Wvh, D_DIM * D_DIM);
  cvt_f32_f16<<<cdiv(D_DIM * D_DIM, 256), 256, 0, stream>>>(Wo, Woh, D_DIM * D_DIM);
  cvt_f32_f16<<<cdiv(FF_DIM * D_DIM, 256), 256, 0, stream>>>(W1, W1h, FF_DIM * D_DIM);
  cvt_f32_f16<<<cdiv(D_DIM * FF_DIM, 256), 256, 0, stream>>>(W2, W2h, D_DIM * FF_DIM);

  const dim3 blk(256);
  const float qscale = 0.17677669529663687f;  // 1/sqrt(HD)

  // --- QKV projections (WMMA), head-split; Q pre-scaled; V transposed ---
  gemm_xwt<2><<<dim3(D_DIM / 64, NTOK / 128), blk, 0, stream>>>(
      xh, Wqh, bq, Qh, NTOK, D_DIM, D_DIM, qscale);
  gemm_xwt<2><<<dim3(D_DIM / 64, NTOK / 128), blk, 0, stream>>>(
      xh, Wkh, bk, Kh, NTOK, D_DIM, D_DIM, 1.0f);
  gemm_xwt<3><<<dim3(D_DIM / 64, NTOK / 128), blk, 0, stream>>>(
      xh, Wvh, bv, Vth, NTOK, D_DIM, D_DIM, 1.0f);

  // --- flash attention (WMMA) ---
  attn_kernel<<<dim3(S_LEN / 128, B_CNT * H_CNT), blk, 0, stream>>>(Qh, Kh, Vth, AOh);

  // --- O projection + residual + LN1 ---
  gemm_xwt<0><<<dim3(D_DIM / 64, NTOK / 128), blk, 0, stream>>>(
      AOh, Woh, bo, Y, NTOK, D_DIM, D_DIM, 1.0f);
  ln_residual<<<NTOK / 8, blk, 0, stream>>>(src, Y, g1, be1, src2, src2h);

  // --- FFN: GELU(src2 @ W1^T + b1) @ W2^T + b2, residual + LN2 ---
  gemm_xwt<1><<<dim3(FF_DIM / 64, NTOK / 128), blk, 0, stream>>>(
      src2h, W1h, b1, Hh, NTOK, FF_DIM, D_DIM, 1.0f);
  gemm_xwt<0><<<dim3(D_DIM / 64, NTOK / 128), blk, 0, stream>>>(
      Hh, W2h, b2, Y, NTOK, D_DIM, FF_DIM, 1.0f);
  ln_residual<<<NTOK / 8, blk, 0, stream>>>(src2, Y, g2, be2, out, nullptr);
}
